// PD_37434934952382
// MI455X (gfx1250) — matrix operand order repaired and checked by reading, not compiled, not run on hardware
//
#include <hip/hip_runtime.h>

// ---------------------------------------------------------------------------
// PCD-align pipeline for MI455X (gfx1250, wave32, WMMA + TDM).
// - All convs + DCN column-GEMM through v_wmma_f32_16x16x32_f16 (fp32 accum).
// - Block-uniform B weight tiles staged into LDS once per block via the
//   Tensor Data Mover (tensor_load_to_lds + s_wait_tensorcnt), consumed with
//   ds_load_b128 by all 4 waves.
// - K loop fully unrolled (compile-time trip count) with ping-pong fragment
//   registers: no inter-buffer v_mov copies, loads issued far ahead of use.
// - Zero padding via a zero guard row (no EXEC churn in the hot loop).
// ---------------------------------------------------------------------------

typedef _Float16 v16h __attribute__((ext_vector_type(16)));
typedef _Float16 v8h  __attribute__((ext_vector_type(8)));
typedef float    v8f  __attribute__((ext_vector_type(8)));
typedef unsigned int v4u __attribute__((ext_vector_type(4)));
typedef int      v4i  __attribute__((ext_vector_type(4)));
typedef int      v8i  __attribute__((ext_vector_type(8)));

// ---------------- layout pack kernels ----------------

__global__ void zero_kernel(_Float16* __restrict__ p, int n) {
  int t = blockIdx.x * 256 + threadIdx.x;
  if (t < n) p[t] = (_Float16)0.f;
}

// NCHW f32 -> NHWC f16 (into a possibly wider channel-concat buffer)
__global__ void pack_in_kernel(const float* __restrict__ in, _Float16* __restrict__ out,
                               int H, int W, int outCs, int outChOff) {
  int t = blockIdx.x * 256 + threadIdx.x;
  int total = 4 * 64 * H * W;
  if (t >= total) return;
  int x = t % W; int r = t / W; int y = r % H; r /= H; int c = r % 64; int b = r / 64;
  out[((long)(b * H + y) * W + x) * outCs + outChOff + c] = (_Float16)in[t];
}

// OIHW f32 conv weight -> B-fragment swizzled f16:
// layout [tap][kchunk32][nchunk16][lane32][elem16], K(e,h)=e+8h+8*(e>>3), N=lane&15
__global__ void pack_wconv_kernel(const float* __restrict__ w, _Float16* __restrict__ wB,
                                  int cin, int coutReal, int coutP) {
  int t = blockIdx.x * 256 + threadIdx.x;
  int nG = coutP >> 4, kG = cin >> 5;
  int total = 9 * kG * nG * 512;
  if (t >= total) return;
  int e = t & 15; int lane = (t >> 4) & 31;
  int r = t >> 9;
  int nc = r % nG; r /= nG;
  int kc = r % kG; int tap = r / kG;
  int o  = nc * 16 + (lane & 15);
  int ci = kc * 32 + e + 8 * (lane >> 4) + ((e >> 3) << 3);
  _Float16 v = (_Float16)0.f;
  if (o < coutReal) v = (_Float16)w[((long)o * cin + ci) * 9 + tap];
  wB[t] = v;
}

// DCN weight (64,64,3,3) -> B-fragment swizzle over K=576 (K index = c*9+j)
__global__ void pack_wcol_kernel(const float* __restrict__ w, _Float16* __restrict__ wB) {
  int t = blockIdx.x * 256 + threadIdx.x;
  const int nG = 4, kG = 18;
  int total = kG * nG * 512;
  if (t >= total) return;
  int e = t & 15; int lane = (t >> 4) & 31;
  int r = t >> 9;
  int nc = r % nG; int kc = r / nG;
  int o = nc * 16 + (lane & 15);
  int k = kc * 32 + e + 8 * (lane >> 4) + ((e >> 3) << 3);
  int c = k / 9, j = k % 9;
  wB[t] = (_Float16)w[((long)o * 64 + c) * 9 + j];
}

// ---------------- TDM staging of the block-uniform B tile -------------------
// 2D tile: rows = taps*cinChunks, each row = 1024 f16 (two N-tiles) = 256 x 8B,
// global row stride = nGrp*512 f16 = nGrp*128 x 8B. Dest: dynamic LDS.
static __device__ __forceinline__ void tdm_stage_B(const _Float16* gsrc, unsigned ldsOff,
                                                   int rows, int rowStride8B) {
  unsigned long long ga = (unsigned long long)(const void*)gsrc;
  v4u g0 = {};
  g0[0] = 1u;                                            // count=1 (valid), user mode
  g0[1] = ldsOff;                                        // lds_addr (bytes)
  g0[2] = (unsigned)ga;                                  // global_addr[31:0]
  g0[3] = (unsigned)((ga >> 32) & 0x1FFFFFFu) | (2u << 30);  // ga[56:32] | type=2
  v8i g1 = {};
  g1[0] = 0x30000;                                       // data_size=3 (8 bytes)
  g1[1] = (int)(256u << 16);                             // tensor_dim0 lo16 (=256)
  g1[2] = (int)((unsigned)rows << 16);                   // tdim0 hi16=0 | tensor_dim1 lo16
  g1[3] = (int)(256u << 16);                             // tdim1 hi16=0 | tile_dim0=256
  g1[4] = rows & 0xFFFF;                                 // tile_dim1 (tile_dim2=0)
  g1[5] = rowStride8B;                                   // tensor_dim0_stride lo32
  g1[6] = 0; g1[7] = 0;
  v4i g2 = {}; v4i g3 = {};
  v8i g4 = {};
  __builtin_amdgcn_tensor_load_to_lds(g0, g1, g2, g3, g4, 0);  // 6-arg form (clang-23)
}

// ---------------- WMMA implicit-GEMM conv / col-GEMM ----------------
// 128 threads = 4 waves; wave = 16 pixels x 32 out channels (2 acc tiles).
// grid.z picks the 32-channel group. TAPS=9: 3x3 conv w/ zero pad via guard
// row; TAPS=1: plain GEMM over Cin=CC*32 (DCN cols). Compile-time CC makes
// the K loop fully unrollable with ping-pong fragments (no register copies).
template <int TAPS, int CC>
__global__ __launch_bounds__(128) void wmma_gemm_kernel(
    const _Float16* __restrict__ in, const _Float16* __restrict__ wB,
    const float* __restrict__ bias, const _Float16* __restrict__ zrow,
    _Float16* __restrict__ outH, float* __restrict__ outF,
    int H, int W,
    int coutReal, int coutP, int outCs, int outChOff,
    int pixOffset, int act) {
  extern __shared__ __attribute__((aligned(64))) _Float16 smem[];

  constexpr int Cin = CC * 32;
  constexpr int TOT = TAPS * CC;        // even for all instantiations used

  const int lane = threadIdx.x & 31;
  const int wv   = threadIdx.x >> 5;
  const int m    = lane & 15;
  const int hlf  = lane >> 4;
  const int pixL = blockIdx.x * 64 + wv * 16 + m;

  int b = 0, y = 0, x = 0;
  if (TAPS == 9) { b = pixL / (H * W); int r = pixL - b * H * W; y = r / W; x = r - y * W; }

  const int nGrp = coutP >> 4;
  const int nc0  = blockIdx.z * 2;

  // --- stage the whole B tile for this block into LDS via the TDM (wave 0) ---
  if (wv == 0) {
    tdm_stage_B(wB + (size_t)nc0 * 512, (unsigned)(size_t)(void*)smem,
                TOT, nGrp * 128);
    __builtin_amdgcn_s_wait_tensorcnt(0);
  }
  __builtin_prefetch(in + (long)pixL * Cin, 0, 1);       // global_prefetch_b8
  __syncthreads();

  v8f acc0 = {}; v8f acc1 = {};

  auto fragAddrA = [&](int kt) -> const _Float16* {
    const int tap = kt / CC;            // compile-time folded after unroll
    const int kc  = kt - tap * CC;
    if (TAPS == 9) {
      int ty = y + tap / 3 - 1, tx = x + tap % 3 - 1;
      bool v = (ty >= 0) && (ty < H) && (tx >= 0) && (tx < W);
      const _Float16* base = v ? in + ((long)(b * H + ty) * W + tx) * Cin : zrow;
      return base + (kc << 5) + (hlf << 3);
    }
    return in + (long)pixL * Cin + (kt << 5) + (hlf << 3);
  };
  auto loadA = [&](int kt) -> v16h {
    const _Float16* ab = fragAddrA(kt);
    v8h lo = *(const v8h*)ab;
    v8h hi = *(const v8h*)(ab + 16);
    v16h a;
#pragma unroll
    for (int i = 0; i < 8; ++i) { a[i] = lo[i]; a[i + 8] = hi[i]; }
    return a;
  };
  auto loadB = [&](int kt, int n) -> v16h {
    return *(const v16h*)(smem + (size_t)kt * 1024 + (n << 9) + (lane << 4));
  };

  // --- fully unrolled ping-pong K loop --------------------------------------
  v16h A0 = loadA(0), B00 = loadB(0, 0), B01 = loadB(0, 1);
  v16h A1 = loadA(1), B10 = loadB(1, 0), B11 = loadB(1, 1);
#pragma unroll
  for (int kt = 0; kt < TOT; kt += 2) {
    acc0 = __builtin_amdgcn_wmma_f32_16x16x32_f16(false, A0, false, B00, (short)0, acc0, false, false);
    acc1 = __builtin_amdgcn_wmma_f32_16x16x32_f16(false, A0, false, B01, (short)0, acc1, false, false);
    if (kt + 2 < TOT) {
      A0 = loadA(kt + 2); B00 = loadB(kt + 2, 0); B01 = loadB(kt + 2, 1);
    }
    acc0 = __builtin_amdgcn_wmma_f32_16x16x32_f16(false, A1, false, B10, (short)0, acc0, false, false);
    acc1 = __builtin_amdgcn_wmma_f32_16x16x32_f16(false, A1, false, B11, (short)0, acc1, false, false);
    if (kt + 3 < TOT) {
      A1 = loadA(kt + 3); B10 = loadB(kt + 3, 0); B11 = loadB(kt + 3, 1);
    }
  }

  // --- epilogue: D lane L, vgpr r -> M = r + 8*(L>>4), N = L&15 -------------
#pragma unroll
  for (int nc = 0; nc < 2; ++nc) {
    v8f A = nc ? acc1 : acc0;
    int ch = (nc0 + nc) * 16 + (lane & 15);
    bool chok = ch < coutReal;
    float bv = chok ? bias[ch] : 0.f;
#pragma unroll
    for (int r = 0; r < 8; ++r) {
      int pl = blockIdx.x * 64 + wv * 16 + r + 8 * hlf;
      int po = pl + pixOffset;
      float v = A[r] + bv;
      if (act) v = (v >= 0.f) ? v : 0.1f * v;
      if (chok) {
        if (outH) outH[(long)po * outCs + outChOff + ch] = (_Float16)v;
        if (outF) {
          int bo = po / (H * W); int r2 = po - bo * H * W;
          int yo = r2 / W; int xo = r2 - yo * W;
          outF[(((long)bo * coutReal + ch) * H + yo) * W + xo] = v;
        }
      }
    }
  }
}

// ---------------- DCNv2 sampling: bilinear gather + sigmoid modulation ------
static __device__ __forceinline__ void corner_acc(
    const _Float16* __restrict__ xin, int xCs, int b, int H, int W, int g,
    int yi, int xi, float wt, float* acc) {
  if (yi < 0 || yi >= H || xi < 0 || xi >= W) return;
  v8h v = *(const v8h*)(xin + ((long)(b * H + yi) * W + xi) * xCs + g * 8);
#pragma unroll
  for (int c = 0; c < 8; ++c) acc[c] += wt * (float)v[c];
}

__global__ void dcn_sample_kernel(const _Float16* __restrict__ xin, int xCs,
                                  const _Float16* __restrict__ om,
                                  _Float16* __restrict__ cols,
                                  int H, int W, int b) {
  int t = blockIdx.x * 256 + threadIdx.x;
  int total = 72 * H * W;
  if (t >= total) return;
  int x = t % W; int r = t / W; int y = r % H; r /= H;
  int j = r % 9; int g = r / 9;
  long obase = ((long)(b * H + y) * W + x) * 224;
  float dy = (float)om[obase + g * 9 + j];
  float dx = (float)om[obase + 72 + g * 9 + j];
  float mr = (float)om[obase + 144 + g * 9 + j];
  float mask = 1.f / (1.f + expf(-mr));
  float py = (float)y + (float)(j / 3 - 1) + dy;
  float px = (float)x + (float)(j % 3 - 1) + dx;
  float y0f = floorf(py), x0f = floorf(px);
  int y0 = (int)y0f, x0 = (int)x0f;
  float wy = py - y0f, wx = px - x0f;
  float acc[8] = {0.f, 0.f, 0.f, 0.f, 0.f, 0.f, 0.f, 0.f};
  corner_acc(xin, xCs, b, H, W, g, y0,     x0,     (1.f - wy) * (1.f - wx), acc);
  corner_acc(xin, xCs, b, H, W, g, y0,     x0 + 1, (1.f - wy) * wx,         acc);
  corner_acc(xin, xCs, b, H, W, g, y0 + 1, x0,     wy * (1.f - wx),         acc);
  corner_acc(xin, xCs, b, H, W, g, y0 + 1, x0 + 1, wy * wx,                 acc);
  long cbase = ((long)y * W + x) * 576 + j;
#pragma unroll
  for (int c = 0; c < 8; ++c)
    cols[cbase + (long)(g * 8 + c) * 9] = (_Float16)(acc[c] * mask);
}

// ---------------- bilinear 2x upsample (half-pixel), fused scale ------------
__global__ void up2_kernel(const _Float16* __restrict__ in, int inCs, int H, int W,
                           _Float16* __restrict__ out, int outCs, int outChOff,
                           float scale) {
  int H2 = 2 * H, W2 = 2 * W;
  int t = blockIdx.x * 256 + threadIdx.x;
  int total = 4 * H2 * W2 * 64;
  if (t >= total) return;
  int c = t & 63; int r = t >> 6;
  int xo = r % W2; r /= W2; int yo = r % H2; int b = r / H2;
  float yi = fminf(fmaxf(0.5f * yo - 0.25f, 0.f), (float)(H - 1));
  float xi = fminf(fmaxf(0.5f * xo - 0.25f, 0.f), (float)(W - 1));
  int y0 = (int)floorf(yi); int x0 = (int)floorf(xi);
  int y1 = min(y0 + 1, H - 1); int x1 = min(x0 + 1, W - 1);
  float wy = yi - (float)y0, wx = xi - (float)x0;
  long b00 = ((long)(b * H + y0) * W + x0) * inCs + c;
  long b01 = ((long)(b * H + y0) * W + x1) * inCs + c;
  long b10 = ((long)(b * H + y1) * W + x0) * inCs + c;
  long b11 = ((long)(b * H + y1) * W + x1) * inCs + c;
  float v = (1.f - wy) * ((1.f - wx) * (float)in[b00] + wx * (float)in[b01])
          +        wy  * ((1.f - wx) * (float)in[b10] + wx * (float)in[b11]);
  out[((long)(b * H2 + yo) * W2 + xo) * outCs + outChOff + c] = (_Float16)(v * scale);
}

// ---------------------------------------------------------------------------
extern "C" void kernel_launch(void* const* d_in, const int* in_sizes, int n_in,
                              void* d_out, int out_size, void* d_ws, size_t ws_size,
                              hipStream_t stream) {
  const int H1 = 160, W1 = 160, H2 = 80, W2 = 80, H3 = 40, W3 = 40;

  const float* fea1 = (const float*)d_in[0];
  const float* fea2 = (const float*)d_in[1];
  const float* fea3 = (const float*)d_in[2];
  const float* w_l3off2 = (const float*)d_in[3];  const float* b_l3off2 = (const float*)d_in[4];
  const float* w_l3om   = (const float*)d_in[5];  const float* b_l3om   = (const float*)d_in[6];
  const float* w_l3dcn  = (const float*)d_in[7];  const float* b_l3dcn  = (const float*)d_in[8];
  const float* w_l2off2 = (const float*)d_in[9];  const float* b_l2off2 = (const float*)d_in[10];
  const float* w_l2off3 = (const float*)d_in[11]; const float* b_l2off3 = (const float*)d_in[12];
  const float* w_l2om   = (const float*)d_in[13]; const float* b_l2om   = (const float*)d_in[14];
  const float* w_l2dcn  = (const float*)d_in[15]; const float* b_l2dcn  = (const float*)d_in[16];
  const float* w_l2fea  = (const float*)d_in[17]; const float* b_l2fea  = (const float*)d_in[18];
  const float* w_l1off2 = (const float*)d_in[19]; const float* b_l1off2 = (const float*)d_in[20];
  const float* w_l1off3 = (const float*)d_in[21]; const float* b_l1off3 = (const float*)d_in[22];
  const float* w_l1om   = (const float*)d_in[23]; const float* b_l1om   = (const float*)d_in[24];
  const float* w_l1dcn  = (const float*)d_in[25]; const float* b_l1dcn  = (const float*)d_in[26];
  const float* w_l1fea  = (const float*)d_in[27]; const float* b_l1fea  = (const float*)d_in[28];

  float* out1 = (float*)d_out;
  float* out2 = out1 + (size_t)4 * 64 * 160 * 160;
  float* out3 = out2 + (size_t)4 * 64 * 80 * 80;

  char* base = (char*)d_ws; size_t off = 0;
  auto alloc = [&](size_t halves) -> _Float16* {
    off = (off + 255) & ~(size_t)255;
    _Float16* p = (_Float16*)(base + off);
    off += halves * 2;
    return p;
  };

  _Float16* zrow    = alloc(640);                       // zero guard row (max Cin+pad)
  _Float16* x3h     = alloc((size_t)4 * H3 * W3 * 64);
  _Float16* l3off   = alloc((size_t)4 * H3 * W3 * 64);
  _Float16* om3     = alloc((size_t)4 * H3 * W3 * 224);
  _Float16* cols3   = alloc((size_t)H3 * W3 * 576);
  _Float16* l3fea   = alloc((size_t)4 * H3 * W3 * 64);
  _Float16* cat2off = alloc((size_t)4 * H2 * W2 * 128);
  _Float16* l2offa  = alloc((size_t)4 * H2 * W2 * 64);
  _Float16* l2off   = alloc((size_t)4 * H2 * W2 * 64);
  _Float16* om2     = alloc((size_t)4 * H2 * W2 * 224);
  _Float16* cols2   = alloc((size_t)H2 * W2 * 576);
  _Float16* cat2fea = alloc((size_t)4 * H2 * W2 * 128);
  _Float16* l2fea   = alloc((size_t)4 * H2 * W2 * 64);
  _Float16* cat1off = alloc((size_t)4 * H1 * W1 * 128);
  _Float16* l1offa  = alloc((size_t)4 * H1 * W1 * 64);
  _Float16* l1off   = alloc((size_t)4 * H1 * W1 * 64);
  _Float16* om1     = alloc((size_t)4 * H1 * W1 * 224);
  _Float16* cols1   = alloc((size_t)H1 * W1 * 576);
  _Float16* cat1fea = alloc((size_t)4 * H1 * W1 * 128);

  zero_kernel<<<3, 256, 0, stream>>>(zrow, 640);

  auto packW = [&](const float* w, int cin, int coutR, int coutP) -> _Float16* {
    size_t n = (size_t)9 * cin * coutP;
    _Float16* p = alloc(n);
    pack_wconv_kernel<<<((int)n + 255) / 256, 256, 0, stream>>>(w, p, cin, coutR, coutP);
    return p;
  };
  auto packWc = [&](const float* w) -> _Float16* {
    size_t n = (size_t)576 * 64;
    _Float16* p = alloc(n);
    pack_wcol_kernel<<<((int)n + 255) / 256, 256, 0, stream>>>(w, p);
    return p;
  };
  _Float16* wb_l3off2 = packW(w_l3off2,  64, 64, 64);
  _Float16* wb_l3om   = packW(w_l3om,    64, 216, 224);
  _Float16* wb_l3dcn  = packWc(w_l3dcn);
  _Float16* wb_l2off2 = packW(w_l2off2, 128, 64, 64);
  _Float16* wb_l2off3 = packW(w_l2off3,  64, 64, 64);
  _Float16* wb_l2om   = packW(w_l2om,    64, 216, 224);
  _Float16* wb_l2dcn  = packWc(w_l2dcn);
  _Float16* wb_l2fea  = packW(w_l2fea,  128, 64, 64);
  _Float16* wb_l1off2 = packW(w_l1off2, 128, 64, 64);
  _Float16* wb_l1off3 = packW(w_l1off3,  64, 64, 64);
  _Float16* wb_l1om   = packW(w_l1om,    64, 216, 224);
  _Float16* wb_l1dcn  = packWc(w_l1dcn);
  _Float16* wb_l1fea  = packW(w_l1fea,  128, 64, 64);

  pack_in_kernel<<<(4 * 64 * H3 * W3 + 255) / 256, 256, 0, stream>>>(fea3, x3h, H3, W3, 64, 0);
  pack_in_kernel<<<(4 * 64 * H2 * W2 + 255) / 256, 256, 0, stream>>>(fea2, cat2off, H2, W2, 128, 0);
  pack_in_kernel<<<(4 * 64 * H1 * W1 + 255) / 256, 256, 0, stream>>>(fea1, cat1off, H1, W1, 128, 0);

  auto conv = [&](const _Float16* in, int Cin, int taps, int H, int W, int Ppix,
                  const _Float16* wB, const float* bias, int coutR, int coutP,
                  _Float16* outH, int outCs, int outChOff, float* outF,
                  int act, int pixOff) {
    dim3 g(Ppix / 64, 1, coutP / 32);
    size_t sh = (size_t)taps * (Cin >> 5) * 2048;   // B tile bytes in LDS
    if (taps == 9 && Cin == 64)
      wmma_gemm_kernel<9, 2><<<g, 128, sh, stream>>>(in, wB, bias, zrow, outH, outF,
                                                     H, W, coutR, coutP, outCs, outChOff, pixOff, act);
    else if (taps == 9 && Cin == 128)
      wmma_gemm_kernel<9, 4><<<g, 128, sh, stream>>>(in, wB, bias, zrow, outH, outF,
                                                     H, W, coutR, coutP, outCs, outChOff, pixOff, act);
    else
      wmma_gemm_kernel<1, 18><<<g, 128, sh, stream>>>(in, wB, bias, zrow, outH, outF,
                                                      H, W, coutR, coutP, outCs, outChOff, pixOff, act);
  };

  // ---------------- L3 ----------------
  conv(x3h,   64, 9, H3, W3, 4 * H3 * W3, wb_l3off2, b_l3off2, 64, 64,  l3off, 64, 0, nullptr, 1, 0);
  conv(l3off, 64, 9, H3, W3, 4 * H3 * W3, wb_l3om,   b_l3om,  216, 224, om3,  224, 0, nullptr, 0, 0);
  for (int b = 0; b < 4; ++b) {
    int tot = 72 * H3 * W3;
    dcn_sample_kernel<<<(tot + 255) / 256, 256, 0, stream>>>(x3h, 64, om3, cols3, H3, W3, b);
    conv(cols3, 576, 1, H3, W3, H3 * W3, wb_l3dcn, b_l3dcn, 64, 64,
         l3fea, 64, 0, out3, 1, b * H3 * W3);
  }

  // ---------------- L2 ----------------
  up2_kernel<<<(4 * H2 * W2 * 64 + 255) / 256, 256, 0, stream>>>(l3off, 64, H3, W3, cat2off, 128, 64, 2.0f);
  conv(cat2off, 128, 9, H2, W2, 4 * H2 * W2, wb_l2off2, b_l2off2, 64, 64, l2offa, 64, 0, nullptr, 1, 0);
  conv(l2offa,   64, 9, H2, W2, 4 * H2 * W2, wb_l2off3, b_l2off3, 64, 64, l2off,  64, 0, nullptr, 1, 0);
  conv(l2off,    64, 9, H2, W2, 4 * H2 * W2, wb_l2om,   b_l2om,  216, 224, om2,  224, 0, nullptr, 0, 0);
  for (int b = 0; b < 4; ++b) {
    int tot = 72 * H2 * W2;
    dcn_sample_kernel<<<(tot + 255) / 256, 256, 0, stream>>>(cat2off, 128, om2, cols2, H2, W2, b);
    conv(cols2, 576, 1, H2, W2, H2 * W2, wb_l2dcn, b_l2dcn, 64, 64,
         cat2fea, 128, 0, nullptr, 0, b * H2 * W2);
  }
  up2_kernel<<<(4 * H2 * W2 * 64 + 255) / 256, 256, 0, stream>>>(l3fea, 64, H3, W3, cat2fea, 128, 64, 1.0f);
  conv(cat2fea, 128, 9, H2, W2, 4 * H2 * W2, wb_l2fea, b_l2fea, 64, 64,
       l2fea, 64, 0, out2, 1, 0);

  // ---------------- L1 ----------------
  up2_kernel<<<(4 * H1 * W1 * 64 + 255) / 256, 256, 0, stream>>>(l2off, 64, H2, W2, cat1off, 128, 64, 2.0f);
  conv(cat1off, 128, 9, H1, W1, 4 * H1 * W1, wb_l1off2, b_l1off2, 64, 64, l1offa, 64, 0, nullptr, 1, 0);
  conv(l1offa,   64, 9, H1, W1, 4 * H1 * W1, wb_l1off3, b_l1off3, 64, 64, l1off,  64, 0, nullptr, 1, 0);
  conv(l1off,    64, 9, H1, W1, 4 * H1 * W1, wb_l1om,   b_l1om,  216, 224, om1,  224, 0, nullptr, 0, 0);
  for (int b = 0; b < 4; ++b) {
    int tot = 72 * H1 * W1;
    dcn_sample_kernel<<<(tot + 255) / 256, 256, 0, stream>>>(cat1off, 128, om1, cols1, H1, W1, b);
    conv(cols1, 576, 1, H1, W1, H1 * W1, wb_l1dcn, b_l1dcn, 64, 64,
         cat1fea, 128, 0, nullptr, 0, b * H1 * W1);
  }
  up2_kernel<<<(4 * H1 * W1 * 64 + 255) / 256, 256, 0, stream>>>(l2fea, 64, H2, W2, cat1fea, 128, 64, 1.0f);
  conv(cat1fea, 128, 9, H1, W1, 4 * H1 * W1, wb_l1fea, b_l1fea, 64, 64,
       nullptr, 0, 0, out1, 0, 0);
}
